// TensPowModel_76914274337118
// MI455X (gfx1250) — matrix operand order repaired and checked by reading, not compiled, not run on hardware
//
#include <hip/hip_runtime.h>
#include <cmath>

// ---------------------------------------------------------------------------
// 64-round ternary-linear chain, fp16, on gfx1250: WMMA + TDM (tensor DMA).
//   h = fmod(h @ W_r^T + b_r, 2);  h = fmod(h + noise_r, 2)
// Each block owns 128 batch rows; each of its 8 waves owns 16 rows. h lives
// in LDS (wave-private slice). All global->LDS staging is done by the Tensor
// Data Mover (wave 0 issues tensor_load_to_lds; TENSORcnt completes in order
// so s_wait_tensorcnt 1 releases the W chunk while the double-buffered noise
// prefetch overlaps the WMMA work). Inner loop: batched B-fragment ds loads,
// early noise loads, then two interleaved v_wmma_f32_16x16x32_f16 chains.
// ---------------------------------------------------------------------------

typedef _Float16 v16h  __attribute__((ext_vector_type(16)));
typedef _Float16 v8h   __attribute__((ext_vector_type(8)));
typedef float    v8f   __attribute__((ext_vector_type(8)));
typedef unsigned int u32x4 __attribute__((ext_vector_type(4)));
typedef unsigned int u32x8 __attribute__((ext_vector_type(8)));

#define HID        256
#define NROUNDS    64
#define BROWS      128      // batch rows per block
#define NTHREADS   256      // 8 wave32
#define RPW        16       // rows per wave
#define WS         272      // sW row stride (halfs): 256 + 16 (TDM pad 8 DW)
#define HS         272      // sH row stride (halfs): 256 + 16 (TDM pad 8 DW)
#define NS         264      // sN row stride (halfs): 256 + 8  (TDM pad 4 DW)
#define NCHUNK     128      // W rows (output cols) staged per LDS chunk

// LDS layout in halfs: sW [NCHUNK][WS] | sH [BROWS][HS] | sN [2][BROWS][NS]
#define SW_OFF     0
#define SH_OFF     (NCHUNK * WS)
#define SN_OFF     (SH_OFF + BROWS * HS)
#define LDS_HALFS  (SN_OFF + 2 * BROWS * NS)          // 137216 halfs = 274432 B

__device__ __forceinline__ float fmod2f(float x) {
    return x - 2.0f * truncf(x * 0.5f);
}

// ---- Tensor Data Mover: 2-D tile load, D# groups 0/1 built in SGPRs -------
// All arguments must be wave-uniform. Dims/strides are in 2-byte elements.
// pad_amount_dw: LDS padding DWORDs inserted after every 128 DWORDs (512 B).
__device__ __forceinline__ void tdm_load_2d(const void* gaddr, unsigned lds_off,
                                            unsigned tensor_d0, unsigned tensor_d1,
                                            unsigned long long d0_stride,
                                            unsigned tile_d0, unsigned tile_d1,
                                            unsigned pad_amount_dw)
{
    const unsigned long long ga = (unsigned long long)gaddr;
    u32x4 g0;
    g0[0] = 1u;                                           // count=1, user D#
    g0[1] = lds_off;                                      // lds_addr [63:32]
    g0[2] = (unsigned)ga;                                 // global_addr[31:0]
    g0[3] = ((unsigned)(ga >> 32) & 0x01FFFFFFu)          // global_addr[56:32]
            | 0x80000000u;                                // type=2 ("image")

    unsigned w0 = (1u << 16);                             // data_size=1 (2 B)
    if (pad_amount_dw)
        w0 |= (1u << 20)                                  // pad_enable
            | (6u << 22)                                  // pad_interval: 128 DW
            | ((pad_amount_dw - 1u) << 25);               // pad_amount
    u32x8 g1;
    g1[0] = w0;                                           // mask=0, no barrier
    g1[1] = (tensor_d0 & 0xFFFFu) << 16;                  // td0[15:0] @ [63:48]
    g1[2] = (tensor_d0 >> 16) | ((tensor_d1 & 0xFFFFu) << 16);
    g1[3] = (tensor_d1 >> 16) | ((tile_d0 & 0xFFFFu) << 16);
    g1[4] = (tile_d1 & 0xFFFFu);                          // tile_dim2 = 0 (2-D)
    g1[5] = (unsigned)d0_stride;                          // dim0_stride[31:0]
    g1[6] = (unsigned)(d0_stride >> 32) & 0xFFFFu;        // dim0_stride[47:32]
    g1[7] = 0u;                                           // dim1_stride unused

    asm volatile("tensor_load_to_lds %0, %1" :: "s"(g0), "s"(g1) : "memory");
}

__device__ __forceinline__ void wait_tensor0() {
    asm volatile("s_wait_tensorcnt 0x0" ::: "memory");
}
__device__ __forceinline__ void wait_tensor1() {
    asm volatile("s_wait_tensorcnt 0x1" ::: "memory");
}

__device__ __forceinline__ unsigned lds_off(const void* p) {
    return (unsigned)(unsigned long long)p;   // aperture: LDS offset is the pointer's low 32 bits
}

__global__ __launch_bounds__(NTHREADS)
void ternary_rounds_kernel(const _Float16* __restrict__ x,
                           const _Float16* __restrict__ noise,
                           const _Float16* __restrict__ weights,
                           const _Float16* __restrict__ bias,
                           _Float16* __restrict__ out)
{
    extern __shared__ _Float16 lds[];
    _Float16* sW = lds + SW_OFF;
    _Float16* sH = lds + SH_OFF;
    _Float16* sN = lds + SN_OFF;             // two buffers of [BROWS][NS]

    const int tid     = threadIdx.x;
    const int lane    = tid & 31;
    const int wave    = tid >> 5;
    const int rowBase = blockIdx.x * BROWS;  // global batch-row base of block
    const int m0      = wave * RPW;          // this wave's local row base
    const int lm      = lane & 15;
    const int hsel    = lane >> 4;           // 0: lanes 0-15, 1: lanes 16-31

    // ---- prologue: TDM-stage x tile -> sH, noise round 0 -> sN[0] ----
    if (wave == 0) {
        tdm_load_2d(x + (size_t)rowBase * HID, lds_off(sH),
                    HID, BROWS, HID, HID, BROWS, /*pad_dw=*/8);       // -> HS
        tdm_load_2d(noise + (size_t)rowBase * NROUNDS * HID, lds_off(sN),
                    HID, BROWS, (unsigned long long)NROUNDS * HID,
                    HID, BROWS, /*pad_dw=*/4);                        // -> NS
    }
    // no wait here: round 0's wait_tensor1 covers these older TDM ops.

    for (int r = 0; r < NROUNDS; ++r) {
        const _Float16* Wr = weights + (size_t)r * HID * HID;

        // ---- round top: everyone done with prev round's sW / old sN buffer
        __syncthreads();
        if (wave == 0) {
            tdm_load_2d(Wr, lds_off(sW), HID, NCHUNK, HID,
                        HID, NCHUNK, /*pad_dw=*/8);                   // W chunk0
            if (r + 1 < NROUNDS) {
                tdm_load_2d(noise + ((size_t)rowBase * NROUNDS + (size_t)(r + 1)) * HID,
                            lds_off(sN + ((r + 1) & 1) * (BROWS * NS)),
                            HID, BROWS, (unsigned long long)NROUNDS * HID,
                            HID, BROWS, /*pad_dw=*/4);                // noise r+1
                wait_tensor1();   // oldest ops (incl. W chunk0) done; noise flies
            } else {
                wait_tensor0();
            }
        }

        // ---- bias row gather for this round (hoisted out of the epilogue) --
        // lane needs cols { t*16 + lm : t = 0..15 }; batched strided u16 loads
        float bv[16];
        #pragma unroll
        for (int t = 0; t < 16; ++t)
            bv[t] = (float)bias[r * HID + t * 16 + lm];

        __syncthreads();      // sW chunk0 (and, round 0, sH/sN[0]) visible

        // ---- load A fragments: this wave's 16 rows x K=256 (ISA 16x32 A layout)
        v16h afrag[8];
        {
            const _Float16* hrow = sH + (m0 + lm) * HS;
            const int t = hsel * 8;
            #pragma unroll
            for (int kk = 0; kk < 8; ++kk) {
                union { v16h v; v8h h[2]; } u;
                u.h[0] = *(const v8h*)(hrow + kk * 32 + t);
                u.h[1] = *(const v8h*)(hrow + kk * 32 + 16 + t);
                afrag[kk] = u.v;
            }
        }

        const _Float16* sNcur = sN + (r & 1) * (BROWS * NS);

        for (int nc = 0; nc < 2; ++nc) {
            if (nc == 1) {
                __syncthreads();                     // chunk0 of sW consumed
                if (wave == 0) {
                    tdm_load_2d(Wr + (size_t)NCHUNK * HID, lds_off(sW),
                                HID, NCHUNK, HID, HID, NCHUNK, /*pad_dw=*/8);
                    wait_tensor0();                  // W chunk1 (+ noise) done
                }
                __syncthreads();
            }

            // ---- 8 output tiles of N=16 within this chunk ----
            #pragma unroll 1
            for (int nt = 0; nt < 8; ++nt) {
                // B-frag: column n = nt*16 + lm (a row of W), table K order:
                // lanes 0-15 -> K=kb..kb+7,kb+16..kb+23; lanes 16-31 -> kb+8..kb+15,kb+24..kb+31
                const _Float16* wrow = sW + (nt * 16 + lm) * WS;
                const int klo = hsel * 8;
                const int col = nc * NCHUNK + nt * 16 + lm;     // 0..255

                // (1) batch all 16 B-fragment ds_load_b128 first
                v16h bfrag[8];
                #pragma unroll
                for (int kk = 0; kk < 8; ++kk) {
                    union { v16h v; v8h h[2]; } b;
                    b.h[0] = *(const v8h*)(wrow + kk * 32 + klo);
                    b.h[1] = *(const v8h*)(wrow + kk * 32 + 16 + klo);
                    bfrag[kk] = b.v;
                }
                // (2) issue noise loads now; DS returns in order, so the WMMA
                //     chain only waits for the B batch while these fly.
                float nv[8];
                #pragma unroll
                for (int v = 0; v < 8; ++v)
                    nv[v] = (float)sNcur[(m0 + v + 8 * hsel) * NS + col];

                // (3) two interleaved accumulation chains (even/odd k)
                v8f acc0 = {}, acc1 = {};
                #pragma unroll
                for (int kk = 0; kk < 4; ++kk) {
                    acc0 = __builtin_amdgcn_wmma_f32_16x16x32_f16(
                        false, afrag[2 * kk],     false, bfrag[2 * kk],
                        (short)0, acc0, false, false);
                    acc1 = __builtin_amdgcn_wmma_f32_16x16x32_f16(
                        false, afrag[2 * kk + 1], false, bfrag[2 * kk + 1],
                        (short)0, acc1, false, false);
                }

                // ---- epilogue: bias + fmod2, + noise + fmod2, f16 -> sH ----
                // D layout: vgpr v, lanes 0-15: M=v ; lanes 16-31: M=v+8
                const float bb = bv[nc * 8 + nt];
                #pragma unroll
                for (int v = 0; v < 8; ++v) {
                    const int rloc = m0 + v + 8 * hsel;
                    float h = (acc0[v] + acc1[v]) + bb;
                    h = fmod2f(h);
                    h += nv[v];
                    h = fmod2f(h);
                    sH[rloc * HS + col] = (_Float16)h;
                }
            }
        }
    }

    // ---- write final h tile to global out ----
    __syncthreads();
    for (int i = tid; i < BROWS * (HID / 8); i += NTHREADS) {
        const int rl = i >> 5;
        const int c8 = (i & 31) * 8;
        *(v8h*)(out + (size_t)(rowBase + rl) * HID + c8) =
            *(const v8h*)(sH + rl * HS + c8);
    }
}

extern "C" void kernel_launch(void* const* d_in, const int* in_sizes, int n_in,
                              void* d_out, int out_size, void* d_ws, size_t ws_size,
                              hipStream_t stream) {
    (void)n_in; (void)out_size; (void)d_ws; (void)ws_size;
    const _Float16* x       = (const _Float16*)d_in[0];   // [16384, 256]
    const _Float16* noise   = (const _Float16*)d_in[1];   // [16384, 64, 256]
    const _Float16* weights = (const _Float16*)d_in[2];   // [64, 256, 256]
    const _Float16* bias    = (const _Float16*)d_in[3];   // [64, 256]
    _Float16*       out     = (_Float16*)d_out;           // [16384, 256]

    const int Bsz    = in_sizes[0] / HID;                 // 16384
    const int nblk   = Bsz / BROWS;                       // 128
    const size_t shm = (size_t)LDS_HALFS * sizeof(_Float16); // 274432 B

    static int shm_configured = 0;  // idempotent host-side attribute set
    if (!shm_configured) {
        (void)hipFuncSetAttribute((const void*)ternary_rounds_kernel,
                                  hipFuncAttributeMaxDynamicSharedMemorySize,
                                  (int)shm);
        shm_configured = 1;
    }

    ternary_rounds_kernel<<<dim3(nblk), dim3(NTHREADS), shm, stream>>>(
        x, noise, weights, bias, out);
}
